// RIMCell_37915971289834
// MI455X (gfx1250) — matrix operand order repaired and checked by reading, not compiled, not run on hardware
//
#include <hip/hip_runtime.h>
#include <math.h>

typedef __bf16 bf16t;
typedef __attribute__((ext_vector_type(16))) __bf16 v16bf;
typedef __attribute__((ext_vector_type(8)))  __bf16 v8bf;
typedef __attribute__((ext_vector_type(8)))  float  v8f;

#define BTOT   8192
#define DDIM   512
#define UNITS  256
#define NRIM   6
#define TOPK   4
#define IVS    400
#define KPAD   416      // 400 padded to multiple of 32
#define HROW   1536     // NRIM*UNITS
#define NTHR   256      // 8 waves per workgroup
#define NWAVE  8

// ---------------------------------------------------------------------------
// WMMA helpers (v_wmma_f32_16x16x32_bf16, wave32)
// ---------------------------------------------------------------------------
__device__ __forceinline__ v8f wmma_bf16(v16bf a, v16bf b, v8f c) {
    return __builtin_amdgcn_wmma_f32_16x16x32_bf16(false, a, false, b, (short)0, c,
                                                   false, false);
}

// A fragment: 16x32 tile (MxK) from row-major src (leading dim ld elements).
// Lane L: row m = L&15, half g = L>>4. Per-lane data = two contiguous 16B runs:
//   f[0..7]  = K(rel) 0..7   at byte offset 0   of (src + m*ld + g*8)
//   f[8..15] = K(rel) 16..23 at byte offset 32
// Requires ld % 8 == 0 (elements) and 16B-aligned base -> two b128 loads.
__device__ __forceinline__ v16bf load_fragA(const bf16t* src, int ld) {
    const int lane = threadIdx.x & 31;
    const int m = lane & 15;
    const int g = lane >> 4;
    const bf16t* p = src + (size_t)m * ld + g * 8;
    v8bf lo = *(const v8bf*)p;
    v8bf hi = *(const v8bf*)(p + 16);
    v16bf f;
#pragma unroll
    for (int e = 0; e < 8; ++e) { f[e] = lo[e]; f[8 + e] = hi[e]; }
    return f;
}

// B fragment: 32x16 tile (KxN) from row-major KxN src (leading dim ld elements).
// Lane L holds column n = L&15, half g = L>>4: f[0..7] = K 8g..8g+7, f[8..15] = K 16+8g..16+8g+7.
__device__ __forceinline__ v16bf load_fragB(const bf16t* src, int ld) {
    const int lane = threadIdx.x & 31;
    const int n = lane & 15;
    const int g = lane >> 4;
    const bf16t* p = src + (size_t)(g * 8) * ld + n;
    v16bf f;
#pragma unroll
    for (int e = 0; e < 8; ++e) { f[e] = p[(size_t)e * ld]; f[8 + e] = p[(size_t)(16 + e) * ld]; }
    return f;
}

// C/D fragment store: lane L -> col n = L&15, VGPR r -> row m = r + 8*(L>>4).
__device__ __forceinline__ void store_cfrag_f32(float* dst, int ld, v8f c) {
    const int lane = threadIdx.x & 31;
    const int n = lane & 15;
    const int g = lane >> 4;
#pragma unroll
    for (int r = 0; r < 8; ++r) dst[(size_t)(r + 8 * g) * ld + n] = c[r];
}

__device__ __forceinline__ void store_cfrag_bf16(bf16t* dst, int ld, v8f c) {
    const int lane = threadIdx.x & 31;
    const int n = lane & 15;
    const int g = lane >> 4;
#pragma unroll
    for (int r = 0; r < 8; ++r) dst[(size_t)(r + 8 * g) * ld + n] = (bf16t)c[r];
}

__device__ __forceinline__ float sigmoidf_(float x) {
    return 1.0f / (1.0f + __expf(-x));
}

// ---------------------------------------------------------------------------
// Weight preparation: fp32 -> bf16, K padded to Kpad (rows >= K zero-filled).
// src: (batch, K, N) row major;  dst: (batch, Kpad, N) row major bf16.
// ---------------------------------------------------------------------------
__global__ void conv_pad_kernel(const float* __restrict__ src, bf16t* __restrict__ dst,
                                int K, int N, int Kpad, int total) {
    int i = blockIdx.x * 256 + threadIdx.x;
    if (i >= total) return;
    int n = i % N;
    int rest = i / N;
    int kp = rest % Kpad;
    int b = rest / Kpad;
    float v = (kp < K) ? src[((size_t)b * K + kp) * N + n] : 0.0f;
    dst[i] = (bf16t)v;
}

// Fold the two value heads: WvEff[d][v] = 0.5*(value_W[d][v] + value_W[d][400+v])
__global__ void conv_wveff_kernel(const float* __restrict__ vW, bf16t* __restrict__ dst) {
    int i = blockIdx.x * 256 + threadIdx.x;
    if (i >= DDIM * IVS) return;
    int v = i % IVS;
    int d = i / IVS;
    dst[i] = (bf16t)(0.5f * (vW[(size_t)d * 800 + v] + vW[(size_t)d * 800 + 400 + v]));
}

// ---------------------------------------------------------------------------
// Stage 1: input attention + top-k mask + LSTM.  grid = B/16, block = 256 (8 waves)
// ---------------------------------------------------------------------------
__global__ __launch_bounds__(NTHR) void rim_stage1_kernel(
    const float* __restrict__ x, const float* __restrict__ hs,
    const float* __restrict__ cs, const float* __restrict__ key_b,
    const float* __restrict__ value_b,
    const bf16t* __restrict__ keyWc,   // (512,128)
    const bf16t* __restrict__ wvEffc,  // (512,400)
    const bf16t* __restrict__ qWc,     // (6,256,128)
    const bf16t* __restrict__ i2hc,    // (6,416,1024)
    const bf16t* __restrict__ h2hc,    // (6,256,1024)
    float* __restrict__ maskOut,       // (B,6)
    float* __restrict__ htOut,         // (B,6,256) f32
    bf16t* __restrict__ htbOut,        // (B,6,256) bf16
    float* __restrict__ cUpdOut)       // d_out region 2
{
    extern __shared__ char smem[];
    bf16t* sX  = (bf16t*)smem;                    // 16 x 512
    bf16t* sH  = sX + 16 * DDIM;                  // [n][m][u]  6x16x256
    bf16t* sRI = sH + NRIM * 16 * UNITS;          // [n][m][k]  6x16x416
    float* sKX = (float*)(sRI + NRIM * 16 * KPAD);// 16 x 128
    float* sQ  = sKX + 16 * 128;                  // [n][m][k]  6x16x128
    float* sV  = sQ + NRIM * 16 * 128;            // 16 x 400
    float* sP  = sV + 16 * IVS;                   // 16 x 6  sigmoid(signal)
    float* sM  = sP + 16 * NRIM;                  // 16 x 6  mask (phase A: raw signal)

    const int b0 = blockIdx.x * 16;
    const int tid = threadIdx.x;
    const int wave = tid >> 5;

    // ---- load + convert inputs to LDS bf16 ----
    for (int i = tid; i < 16 * DDIM; i += NTHR) {
        int m = i >> 9, d = i & 511;
        sX[i] = (bf16t)x[(size_t)(b0 + m) * DDIM + d];
    }
    for (int i = tid; i < NRIM * 16 * UNITS; i += NTHR) {
        int n = i / (16 * UNITS);
        int r = i % (16 * UNITS);
        int m = r >> 8, u = r & 255;
        sH[i] = (bf16t)hs[(size_t)(b0 + m) * HROW + n * UNITS + u];
    }
    __syncthreads();

    // ---- WMMA jobs: 0..7 kx (16x128,K=512), 8..32 vmean (16x400,K=512),
    //                 33..80 q per rim (16x128,K=256) ----
    for (int job = wave; job < 81; job += NWAVE) {
        v8f acc = {0.f, 0.f, 0.f, 0.f, 0.f, 0.f, 0.f, 0.f};
        if (job < 8) {
            const int c = job;
            for (int kc = 0; kc < 16; ++kc) {
                v16bf a = load_fragA(sX + kc * 32, DDIM);
                v16bf b = load_fragB(keyWc + (size_t)(kc * 32) * 128 + c * 16, 128);
                acc = wmma_bf16(a, b, acc);
            }
            store_cfrag_f32(sKX + c * 16, 128, acc);
        } else if (job < 33) {
            const int c = job - 8;
            for (int kc = 0; kc < 16; ++kc) {
                v16bf a = load_fragA(sX + kc * 32, DDIM);
                v16bf b = load_fragB(wvEffc + (size_t)(kc * 32) * IVS + c * 16, IVS);
                acc = wmma_bf16(a, b, acc);
            }
            store_cfrag_f32(sV + c * 16, IVS, acc);
        } else {
            const int j = job - 33;
            const int n = j >> 3, c = j & 7;
            for (int kc = 0; kc < 8; ++kc) {
                v16bf a = load_fragA(sH + (size_t)n * 16 * UNITS + kc * 32, UNITS);
                v16bf b = load_fragB(qWc + ((size_t)n * UNITS + kc * 32) * 128 + c * 16, 128);
                acc = wmma_bf16(a, b, acc);
            }
            store_cfrag_f32(sQ + (size_t)n * 16 * 128 + c * 16, 128, acc);
        }
    }
    __syncthreads();

    // ---- bias add (key_b to kx, folded value_b to vmean) ----
    for (int i = tid; i < 16 * 128; i += NTHR) sKX[i] += key_b[i & 127];
    for (int i = tid; i < 16 * IVS; i += NTHR) {
        int v = i % IVS;
        sV[i] += 0.5f * (value_b[v] + value_b[400 + v]);
    }
    __syncthreads();

    // ---- signal = (q . kx)/16 ; p = sigmoid(signal) ----
    if (tid < 96) {
        const int m = tid / NRIM, n = tid % NRIM;
        float s = 0.f;
        for (int k = 0; k < 128; ++k)
            s += sQ[(size_t)n * 2048 + m * 128 + k] * sKX[m * 128 + k];
        s *= (1.0f / 16.0f);
        sP[m * NRIM + n] = sigmoidf_(s);
        sM[m * NRIM + n] = s;  // raw signal for top-k
    }
    __syncthreads();

    // ---- top-4 of 6 (strict >, lower index wins ties, like lax.top_k) ----
    if (tid < 16) {
        const int m = tid;
        float sig[NRIM];
        bool used[NRIM];
        for (int n = 0; n < NRIM; ++n) { sig[n] = sM[m * NRIM + n]; used[n] = false; }
        for (int t = 0; t < TOPK; ++t) {
            int best = 0; float bv = -3.4e38f;
            for (int n = 0; n < NRIM; ++n)
                if (!used[n] && sig[n] > bv) { bv = sig[n]; best = n; }
            used[best] = true;
        }
        for (int n = 0; n < NRIM; ++n) {
            float mk = used[n] ? 1.0f : 0.0f;
            sM[m * NRIM + n] = mk;
            maskOut[(size_t)(b0 + m) * NRIM + n] = mk;
        }
    }
    __syncthreads();

    // ---- rnn_in[n][m][k] = p * mask * vmean  (bf16, K-padded) ----
    for (int i = tid; i < NRIM * 16 * KPAD; i += NTHR) {
        int n = i / (16 * KPAD);
        int r = i % (16 * KPAD);
        int m = r / KPAD, k = r % KPAD;
        float val = 0.f;
        if (k < IVS) val = sP[m * NRIM + n] * sM[m * NRIM + n] * sV[m * IVS + k];
        sRI[i] = (bf16t)val;
    }
    __syncthreads();

    // ---- preact + LSTM: jobs over (rim n, 16-col block c of UNITS) ----
    for (int job = wave; job < NRIM * 16; job += NWAVE) {
        const int n = job >> 4, c = job & 15;
        v8f acc[4];
#pragma unroll
        for (int gI = 0; gI < 4; ++gI) acc[gI] = (v8f){0.f,0.f,0.f,0.f,0.f,0.f,0.f,0.f};
        for (int kc = 0; kc < KPAD / 32; ++kc) {       // rnn_in @ i2h
            v16bf a = load_fragA(sRI + (size_t)n * 16 * KPAD + kc * 32, KPAD);
#pragma unroll
            for (int gI = 0; gI < 4; ++gI) {
                v16bf b = load_fragB(i2hc + ((size_t)n * KPAD + kc * 32) * 1024 +
                                         gI * UNITS + c * 16, 1024);
                acc[gI] = wmma_bf16(a, b, acc[gI]);
            }
        }
        for (int kc = 0; kc < UNITS / 32; ++kc) {      // hs @ h2h
            v16bf a = load_fragA(sH + (size_t)n * 16 * UNITS + kc * 32, UNITS);
#pragma unroll
            for (int gI = 0; gI < 4; ++gI) {
                v16bf b = load_fragB(h2hc + ((size_t)n * UNITS + kc * 32) * 1024 +
                                         gI * UNITS + c * 16, 1024);
                acc[gI] = wmma_bf16(a, b, acc[gI]);
            }
        }
        const int lane = tid & 31;
        const int col = c * 16 + (lane & 15);
        const int g = lane >> 4;
#pragma unroll
        for (int r = 0; r < 8; ++r) {
            const int m = r + 8 * g;
            float nc = tanhf(acc[0][r]);
            float ig = sigmoidf_(acc[1][r]);
            float fg = sigmoidf_(acc[2][r]);
            float og = sigmoidf_(acc[3][r]);
            const size_t idx = (size_t)(b0 + m) * HROW + n * UNITS + col;
            float csv = cs[idx];
            float ct = csv * fg + ig * nc;
            float ht = og * tanhf(ct);
            float mk = sM[m * NRIM + n];
            cUpdOut[idx] = ct * mk + csv * (1.0f - mk);
            htOut[idx] = ht;
            htbOut[idx] = (bf16t)ht;
        }
    }
}

// ---------------------------------------------------------------------------
// Stage 2: communication attention + output.  grid = B/16, block = 256 (8 waves)
// ---------------------------------------------------------------------------
__global__ __launch_bounds__(NTHR) void rim_comm_kernel(
    const float* __restrict__ hs, const float* __restrict__ maskp,
    const float* __restrict__ ht, const bf16t* __restrict__ htb,
    const bf16t* __restrict__ wkC,   // (6,256,128)
    const bf16t* __restrict__ wqC,   // (6,256,128)
    const bf16t* __restrict__ wvC,   // (6,256,400)
    const bf16t* __restrict__ woutC, // (6,416,256)
    float* __restrict__ out0, float* __restrict__ out1)
{
    extern __shared__ char smem[];
    float* sCK = (float*)smem;                     // [n][m][k] 6x16x128
    float* sCQ = sCK + NRIM * 16 * 128;            // [n][m][k] 6x16x128
    bf16t* sCV = (bf16t*)(sCQ + NRIM * 16 * 128);  // [n][m][v] 6x16x400
    bf16t* sCT = sCV + NRIM * 16 * IVS;            // [n][m][v] 6x16x416

    const int b0 = blockIdx.x * 16;
    const int tid = threadIdx.x;
    const int wave = tid >> 5;
    const bf16t* hbase = htb + (size_t)b0 * HROW;

    // ---- projections via WMMA: ck 48 jobs, cq 48 jobs, cv 150 jobs ----
    for (int job = wave; job < 246; job += NWAVE) {
        v8f acc = {0.f, 0.f, 0.f, 0.f, 0.f, 0.f, 0.f, 0.f};
        if (job < 96) {
            const int isQ = (job >= 48);
            const int j = isQ ? job - 48 : job;
            const int n = j >> 3, c = j & 7;
            const bf16t* w = isQ ? wqC : wkC;
            for (int kc = 0; kc < 8; ++kc) {
                v16bf a = load_fragA(hbase + n * UNITS + kc * 32, HROW);
                v16bf b = load_fragB(w + ((size_t)n * UNITS + kc * 32) * 128 + c * 16, 128);
                acc = wmma_bf16(a, b, acc);
            }
            store_cfrag_f32((isQ ? sCQ : sCK) + (size_t)n * 16 * 128 + c * 16, 128, acc);
        } else {
            const int j = job - 96;
            const int n = j / 25, c = j % 25;
            for (int kc = 0; kc < 8; ++kc) {
                v16bf a = load_fragA(hbase + n * UNITS + kc * 32, HROW);
                v16bf b = load_fragB(wvC + ((size_t)n * UNITS + kc * 32) * IVS + c * 16, IVS);
                acc = wmma_bf16(a, b, acc);
            }
            store_cfrag_bf16(sCV + (size_t)n * 16 * IVS + c * 16, IVS, acc);
        }
    }
    __syncthreads();

    // ---- per (row m, head h): 6x6 scores, softmax, masked probs, ctx ----
    if (tid < 64) {
        const int m = tid >> 2;
        const int h = tid & 3;
        float mk[NRIM];
        for (int n = 0; n < NRIM; ++n) mk[n] = maskp[(size_t)(b0 + m) * NRIM + n];
        float sc[NRIM][NRIM];
        const float inv = 0.1767766952966369f;  // 1/sqrt(32)
        for (int nq = 0; nq < NRIM; ++nq)
            for (int nk = 0; nk < NRIM; ++nk) {
                float s = 0.f;
                for (int k = 0; k < 32; ++k)
                    s += sCQ[(size_t)nq * 2048 + m * 128 + h * 32 + k] *
                         sCK[(size_t)nk * 2048 + m * 128 + h * 32 + k];
                sc[nq][nk] = s * inv;
            }
        for (int nq = 0; nq < NRIM; ++nq) {
            float mx = sc[nq][0];
            for (int nk = 1; nk < NRIM; ++nk) mx = fmaxf(mx, sc[nq][nk]);
            float den = 0.f;
            for (int nk = 0; nk < NRIM; ++nk) {
                sc[nq][nk] = __expf(sc[nq][nk] - mx);
                den += sc[nq][nk];
            }
            float scale = mk[nq] / den;
            for (int nk = 0; nk < NRIM; ++nk) sc[nq][nk] *= scale;
        }
        for (int nq = 0; nq < NRIM; ++nq)
            for (int v = 0; v < 100; ++v) {
                float s = 0.f;
                for (int nk = 0; nk < NRIM; ++nk)
                    s += sc[nq][nk] * (float)sCV[(size_t)nk * 6400 + m * IVS + h * 100 + v];
                sCT[(size_t)nq * 16 * KPAD + m * KPAD + h * 100 + v] = (bf16t)s;
            }
    }
    // zero the K padding of ctx
    for (int i = tid; i < NRIM * 16 * 16; i += NTHR) {
        int n = i >> 8;
        int r = i & 255;
        int m = r >> 4, k = IVS + (r & 15);
        sCT[(size_t)n * 16 * KPAD + m * KPAD + k] = (bf16t)0.0f;
    }
    __syncthreads();

    // ---- h_comm = ctx @ Wout + h_t ; blend and write h_upd twice ----
    for (int job = wave; job < NRIM * 16; job += NWAVE) {
        const int n = job >> 4, c = job & 15;
        v8f acc = {0.f, 0.f, 0.f, 0.f, 0.f, 0.f, 0.f, 0.f};
        for (int kc = 0; kc < KPAD / 32; ++kc) {
            v16bf a = load_fragA(sCT + (size_t)n * 16 * KPAD + kc * 32, KPAD);
            v16bf b = load_fragB(woutC + ((size_t)n * KPAD + kc * 32) * UNITS + c * 16, UNITS);
            acc = wmma_bf16(a, b, acc);
        }
        const int lane = tid & 31;
        const int u = c * 16 + (lane & 15);
        const int g = lane >> 4;
#pragma unroll
        for (int r = 0; r < 8; ++r) {
            const int m = r + 8 * g;
            const size_t idx = (size_t)(b0 + m) * HROW + n * UNITS + u;
            float mk = maskp[(size_t)(b0 + m) * NRIM + n];
            float hcomm = acc[r] + ht[idx];
            float val = (mk > 0.5f) ? hcomm : hs[idx];
            out0[idx] = val;
            out1[idx] = val;
        }
    }
}

// ---------------------------------------------------------------------------
// Host launcher
// ---------------------------------------------------------------------------
extern "C" void kernel_launch(void* const* d_in, const int* in_sizes, int n_in,
                              void* d_out, int out_size, void* d_ws, size_t ws_size,
                              hipStream_t stream) {
    const float* x       = (const float*)d_in[0];
    const float* hs      = (const float*)d_in[1];
    const float* cs      = (const float*)d_in[2];
    const float* key_W   = (const float*)d_in[3];
    const float* key_b   = (const float*)d_in[4];
    const float* value_W = (const float*)d_in[5];
    const float* value_b = (const float*)d_in[6];
    const float* query_W = (const float*)d_in[7];
    const float* i2h     = (const float*)d_in[8];
    const float* h2h     = (const float*)d_in[9];
    const float* Wk      = (const float*)d_in[10];
    const float* Wv      = (const float*)d_in[11];
    const float* Wq      = (const float*)d_in[12];
    const float* Wout    = (const float*)d_in[13];
    float* out = (float*)d_out;
    (void)in_sizes; (void)n_in; (void)out_size; (void)ws_size;

    char* ws = (char*)d_ws;
    size_t o = 0;
    auto carve = [&](size_t bytes) -> void* {
        void* p = ws + o;
        o = (o + bytes + 255) & ~(size_t)255;
        return p;
    };
    bf16t* keyWc  = (bf16t*)carve((size_t)DDIM * 128 * 2);
    bf16t* wvEffc = (bf16t*)carve((size_t)DDIM * IVS * 2);
    bf16t* qWc    = (bf16t*)carve((size_t)NRIM * UNITS * 128 * 2);
    bf16t* i2hc   = (bf16t*)carve((size_t)NRIM * KPAD * 1024 * 2);
    bf16t* h2hc   = (bf16t*)carve((size_t)NRIM * UNITS * 1024 * 2);
    bf16t* wkC    = (bf16t*)carve((size_t)NRIM * UNITS * 128 * 2);
    bf16t* wqC    = (bf16t*)carve((size_t)NRIM * UNITS * 128 * 2);
    bf16t* wvC    = (bf16t*)carve((size_t)NRIM * UNITS * IVS * 2);
    bf16t* woutC  = (bf16t*)carve((size_t)NRIM * KPAD * UNITS * 2);
    float* maskWs = (float*)carve((size_t)BTOT * NRIM * 4);
    float* htWs   = (float*)carve((size_t)BTOT * HROW * 4);
    bf16t* htbWs  = (bf16t*)carve((size_t)BTOT * HROW * 2);

    auto convT = [&](const float* src, bf16t* dst, int K, int N, int batch, int Kpad) {
        int total = batch * N * Kpad;
        conv_pad_kernel<<<(total + 255) / 256, 256, 0, stream>>>(src, dst, K, N, Kpad, total);
    };
    convT(key_W,   keyWc, DDIM,  128,  1,    DDIM);
    convT(query_W, qWc,   UNITS, 128,  NRIM, UNITS);
    convT(i2h,     i2hc,  IVS,   1024, NRIM, KPAD);
    convT(h2h,     h2hc,  UNITS, 1024, NRIM, UNITS);
    convT(Wk,      wkC,   UNITS, 128,  NRIM, UNITS);
    convT(Wq,      wqC,   UNITS, 128,  NRIM, UNITS);
    convT(Wv,      wvC,   UNITS, IVS,  NRIM, UNITS);
    convT(Wout,    woutC, IVS,   UNITS, NRIM, KPAD);
    {
        int total = DDIM * IVS;
        conv_wveff_kernel<<<(total + 255) / 256, 256, 0, stream>>>(value_W, wvEffc);
    }

    const size_t hsz = (size_t)BTOT * HROW;
    float* out0 = out;
    float* out1 = out + hsz;
    float* out2 = out + 2 * hsz;

    const int smem1 = 16 * DDIM * 2 + NRIM * 16 * UNITS * 2 + NRIM * 16 * KPAD * 2 +
                      16 * 128 * 4 + NRIM * 16 * 128 * 4 + 16 * IVS * 4 +
                      16 * NRIM * 4 + 16 * NRIM * 4;                       // ~229 KB
    const int smem2 = NRIM * 16 * 128 * 4 * 2 + NRIM * 16 * IVS * 2 +
                      NRIM * 16 * KPAD * 2;                                // ~255 KB
    (void)hipFuncSetAttribute((const void*)rim_stage1_kernel,
                              hipFuncAttributeMaxDynamicSharedMemorySize, smem1);
    (void)hipFuncSetAttribute((const void*)rim_comm_kernel,
                              hipFuncAttributeMaxDynamicSharedMemorySize, smem2);

    rim_stage1_kernel<<<BTOT / 16, NTHR, smem1, stream>>>(
        x, hs, cs, key_b, value_b, keyWc, wvEffc, qWc, i2hc, h2hc,
        maskWs, htWs, htbWs, out2);

    rim_comm_kernel<<<BTOT / 16, NTHR, smem2, stream>>>(
        hs, maskWs, htWs, htbWs, wkC, wqC, wvC, woutC, out0, out1);
}
